// OURLSTM_20968030339130
// MI455X (gfx1250) — matrix operand order repaired and checked by reading, not compiled, not run on hardware
//
#include <hip/hip_runtime.h>

#define N_NODES 50000
#define N_EDGES 800000
#define F_IN    32
#define HID     64
#define NC      192          // 3 live gates (i, c, o) * 64
#define KTOT    128          // 4 Chebyshev orders * 32 features

typedef __attribute__((ext_vector_type(2)))  float    v2f;
typedef __attribute__((ext_vector_type(8)))  float    v8f;
typedef __attribute__((ext_vector_type(16))) _Float16 v16h;

// workspace layout (float element offsets, all 64B aligned)
#define DEG_OFF   0u         // 50000  (deg, then d^-1/2 in place)
#define NORM_OFF  50048u     // 800000
#define TCAT_OFF  850048u    // 50000*128  (T0|T1|T2|T3 interleaved per row)
#define WCAT_OFF  7250048u   // 128*192
#define G_OFF     7274624u   // 50000*192

__device__ __forceinline__ float sigmoidf(float x) { return 1.0f / (1.0f + expf(-x)); }

__global__ void k_zero(float* __restrict__ p, int n) {
    int i = blockIdx.x * blockDim.x + threadIdx.x;
    if (i < n) p[i] = 0.0f;
}

__global__ void k_deg(const int* __restrict__ ei, const float* __restrict__ w,
                      float* __restrict__ deg) {
    int e = blockIdx.x * blockDim.x + threadIdx.x;
    if (e < N_EDGES) atomicAdd(&deg[ei[e]], w[e]);   // segment_sum over src
}

__global__ void k_dinv(float* __restrict__ deg) {
    int n = blockIdx.x * blockDim.x + threadIdx.x;
    if (n < N_NODES) {
        float d = deg[n];
        deg[n] = (d > 0.0f) ? (1.0f / sqrtf(fmaxf(d, 1e-30f))) : 0.0f;
    }
}

__global__ void k_norm(const int* __restrict__ ei, const float* __restrict__ w,
                       const float* __restrict__ dinv, float* __restrict__ nv) {
    int e = blockIdx.x * blockDim.x + threadIdx.x;
    if (e < N_EDGES) nv[e] = -dinv[ei[e]] * w[e] * dinv[ei[N_EDGES + e]];
}

// Tcat row layout: [T0(0..31) | T1(32..63) | T2(64..95) | T3(96..127)]
__global__ void k_initT(const float* __restrict__ x, float* __restrict__ T) {
    int i = blockIdx.x * blockDim.x + threadIdx.x;
    if (i < N_NODES * F_IN) {
        int n = i >> 5, f = i & 31;
        float v = x[i];
        T[n * KTOT +      f] = v;     // T0 = x
        T[n * KTOT + 32 + f] = 0.0f;  // T1 accumulator
        T[n * KTOT + 64 + f] = -v;    // T2 = 2*prop(T1) - T0
    }
}

__global__ void k_init3(float* __restrict__ T) {
    int i = blockIdx.x * blockDim.x + threadIdx.x;
    if (i < N_NODES * F_IN) {
        int n = i >> 5, f = i & 31;
        T[n * KTOT + 96 + f] = -T[n * KTOT + 32 + f];  // T3 = 2*prop(T2) - T1
    }
}

// scatter: Tcat[dst, slotDst+f] += scale * norm[e] * Tcat[src, slotSrc+f]
__global__ void k_scatter(const int* __restrict__ ei, const float* __restrict__ nv,
                          float* __restrict__ T, int slotSrc, int slotDst, float scale) {
    long long tid = (long long)blockIdx.x * blockDim.x + threadIdx.x;
    if (tid >= (long long)N_EDGES * 8) return;
    int e = (int)(tid >> 3);
    int q = ((int)tid & 7) * 4;
    int src = ei[e], dst = ei[N_EDGES + e];
    float c = scale * nv[e];
    const float4 t = *(const float4*)&T[src * KTOT + slotSrc + q];
    float* o = &T[dst * KTOT + slotDst + q];
    atomicAdd(o + 0, c * t.x);
    atomicAdd(o + 1, c * t.y);
    atomicAdd(o + 2, c * t.z);
    atomicAdd(o + 3, c * t.w);
}

// Wcat[k*32+i][go*64+j] = Wx[gate(go)][k][i][j], gate map {0:i, 1:c, 2:o}->{0,2,3}
__global__ void k_wcat(const float* __restrict__ Wx, float* __restrict__ Wc) {
    int i = blockIdx.x * blockDim.x + threadIdx.x;
    if (i < KTOT * NC) {
        int kidx = i / NC, c = i % NC;
        int go = c >> 6, j = c & 63;
        int gate = (go == 0) ? 0 : (go == 1 ? 2 : 3);
        int k = kidx >> 5, fi = kidx & 31;
        Wc[i] = Wx[gate * (4 * F_IN * HID) + k * (F_IN * HID) + fi * HID + j];
    }
}

// G[50000 x 192] = Tcat[50000 x 128] @ Wcat[128 x 192], f32 WMMA
__global__ __launch_bounds__(256) void k_gemm(const float* __restrict__ T,
                                              const float* __restrict__ W,
                                              float* __restrict__ G) {
    const int lane  = threadIdx.x & 31;
    const int wave  = threadIdx.x >> 5;
    const int NTN   = NC / 16;                     // 12
    const int NT    = (N_NODES / 16) * NTN;        // 37500
    int tile = blockIdx.x * 8 + wave;
    if (tile >= NT) return;
    const int mT = tile / NTN, nT = tile % NTN;
    const int r     = lane & 15;
    const int khalf = lane >> 4;

    v8f acc = {};
#if defined(__has_builtin) && __has_builtin(__builtin_amdgcn_wmma_f32_16x16x4_f32)
    // fp32 path: A 16x4 (lane=row, khalf selects K pair), B 4x16 mirrored
    const float* arow = T + (mT * 16 + r) * KTOT + khalf * 2;
    const float* bcol = W + (khalf * 2) * NC + nT * 16 + r;
#pragma unroll
    for (int kk = 0; kk < 32; ++kk) {
        v2f a = *(const v2f*)(arow + kk * 4);
        v2f b;
        b.x = bcol[(kk * 4) * NC];
        b.y = bcol[(kk * 4 + 1) * NC];
        acc = __builtin_amdgcn_wmma_f32_16x16x4_f32(false, a, false, b,
                                                    (short)0, acc, false, false);
    }
#else
    // fallback: f16 WMMA 16x16x32 (codegen-confirmed builtin), fp32 accumulate
    const float* arow = T + (mT * 16 + r) * KTOT;
#pragma unroll
    for (int kc = 0; kc < 4; ++kc) {
        v16h a, b;
#pragma unroll
        for (int hi = 0; hi < 16; ++hi) {
            int kA = kc * 32 + ((hi >> 3) << 4) + khalf * 8 + (hi & 7);
            a[hi] = (_Float16)arow[kA];
            int kB = kc * 32 + ((hi >> 3) << 4) + khalf * 8 + (hi & 7);
            b[hi] = (_Float16)W[kB * NC + nT * 16 + r];
        }
        acc = __builtin_amdgcn_wmma_f32_16x16x32_f16(false, a, false, b,
                                                     (short)0, acc, false, false);
    }
#endif
    float* g = G + (mT * 16 + khalf * 8) * NC + nT * 16 + r;
#pragma unroll
    for (int v = 0; v < 8; ++v) g[v * NC] = acc[v];
}

// gates + peephole + tanh + linear head; 4 nodes / 256-thread block
__global__ __launch_bounds__(256) void k_epi(const float* __restrict__ G,
                                             const float* __restrict__ bx,
                                             const float* __restrict__ bh,
                                             const float* __restrict__ bg,
                                             const float* __restrict__ wc,
                                             const float* __restrict__ lw,
                                             const float* __restrict__ lb,
                                             float* __restrict__ out) {
    __shared__ float s[256];
    int tid  = threadIdx.x;
    int node = blockIdx.x * 4 + (tid >> 6);
    int j    = tid & 63;
    const float* g = G + node * NC;
    float bi = bx[0 * HID + j] + bh[0 * HID + j] + bg[0 * HID + j];
    float bc = bx[2 * HID + j] + bh[2 * HID + j] + bg[2 * HID + j];
    float bo = bx[3 * HID + j] + bh[3 * HID + j] + bg[3 * HID + j];
    float I  = sigmoidf(g[0 * HID + j] + bi);
    float Tc = tanhf   (g[1 * HID + j] + bc);
    float C  = I * Tc;                                   // F*C_prev == 0
    float O  = sigmoidf(g[2 * HID + j] + bo + wc[2 * HID + j] * C);
    float H  = O * tanhf(C);
    s[tid] = H * lw[j];
    __syncthreads();
    for (int off = 32; off > 0; off >>= 1) {
        if (j < off) s[tid] += s[tid + off];
        __syncthreads();
    }
    if (j == 0) out[node] = s[tid] + lb[0];
}

extern "C" void kernel_launch(void* const* d_in, const int* in_sizes, int n_in,
                              void* d_out, int out_size, void* d_ws, size_t ws_size,
                              hipStream_t stream) {
    const float* x   = (const float*)d_in[0];
    const int*   ei  = (const int*)  d_in[1];   // edge_index [2, E]
    const float* w   = (const float*)d_in[2];
    const float* Wx  = (const float*)d_in[3];
    const float* bx  = (const float*)d_in[4];
    /* Wh = d_in[5] unused: ChebConv(H=0) == bias */
    const float* bh  = (const float*)d_in[6];
    const float* wc  = (const float*)d_in[7];
    const float* bg  = (const float*)d_in[8];
    const float* lw  = (const float*)d_in[9];
    const float* lb  = (const float*)d_in[10];

    float* ws  = (float*)d_ws;
    float* deg = ws + DEG_OFF;
    float* nv  = ws + NORM_OFF;
    float* T   = ws + TCAT_OFF;
    float* Wc  = ws + WCAT_OFF;
    float* G   = ws + G_OFF;
    float* out = (float*)d_out;

    const int B = 256;
    k_zero <<<(N_NODES + B - 1) / B, B, 0, stream>>>(deg, N_NODES);
    k_deg  <<<(N_EDGES + B - 1) / B, B, 0, stream>>>(ei, w, deg);
    k_dinv <<<(N_NODES + B - 1) / B, B, 0, stream>>>(deg);
    k_norm <<<(N_EDGES + B - 1) / B, B, 0, stream>>>(ei, w, deg, nv);
    k_initT<<<(N_NODES * F_IN + B - 1) / B, B, 0, stream>>>(x, T);

    const int SG = (int)(((long long)N_EDGES * 8 + B - 1) / B);
    k_scatter<<<SG, B, 0, stream>>>(ei, nv, T,  0, 32, 1.0f);  // T1 = prop(T0)
    k_init3  <<<(N_NODES * F_IN + B - 1) / B, B, 0, stream>>>(T);
    k_scatter<<<SG, B, 0, stream>>>(ei, nv, T, 32, 64, 2.0f);  // T2 += 2*prop(T1)
    k_scatter<<<SG, B, 0, stream>>>(ei, nv, T, 64, 96, 2.0f);  // T3 += 2*prop(T2)

    k_wcat<<<(KTOT * NC + B - 1) / B, B, 0, stream>>>(Wx, Wc);

    const int NT = (N_NODES / 16) * (NC / 16);                 // 37500 tiles
    k_gemm<<<(NT + 7) / 8, 256, 0, stream>>>(T, Wc, G);

    k_epi<<<N_NODES / 4, 256, 0, stream>>>(G, bx, bh, bg, wc, lw, lb, out);
}
